// ActorNet_88708254531742
// MI455X (gfx1250) — matrix operand order repaired and checked by reading, not compiled, not run on hardware
//
#include <hip/hip_runtime.h>
#include <hip/hip_fp16.h>

// ---------------------------------------------------------------------------
// ActorNet (encoder MLP -> 2-layer LSTM over T=512 -> policy/baseline heads)
// MI455X / gfx1250, wave32, f16 WMMA (v_wmma_f32_16x16x32_f16) everywhere.
// Round 4: LSTM weights resident in LDS (274KB of the WGP's 320KB) -> the
// recurrent GEMMs read B-fragments via ds_read_b128 with zero per-step L2
// weight traffic (was 8x-redundant per WG).  Bank-conflict-free row pad.
// ---------------------------------------------------------------------------

typedef _Float16 h16;
typedef __attribute__((ext_vector_type(8)))  _Float16 v8h;
typedef __attribute__((ext_vector_type(16))) _Float16 v16h;
typedef __attribute__((ext_vector_type(8)))  float    v8f;

#define T_STEPS 512
#define BATCH   128
#define SDIM    128
#define NDIM    512
#define ADIM    30
#define CORE_R  543
#define CORE_P  544            // padded to 34*16
#define G4P     2176           // 4*CORE_P
#define KCAT    1088           // 2*CORE_P (concat [x | h] along K)
#define ROWS    (T_STEPS*BATCH)   // 65536
#define NWG_LSTM 34            // CORE_P/16 workgroups own 16 gate-cols each
#define LDS_ROW  1096          // 1088 + 8 pad: stride 548 dw == 36 mod 64 banks
                               // -> 16 lanes/phase hit disjoint 4-bank groups

// ---------------- fragment helpers (layouts per CDNA5 ISA 7.12.2) ----------
// A 16x32 f16: lane L (m=L%16, h=L/16) holds K = {8h..8h+7} U {16+8h..23+8h}
// B 32x16 f16: lane L (n=L%16, h=L/16) holds K = {8h..8h+7} U {16+8h..23+8h} of column n
// D 16x16 f32: VGPR r, lane L -> m = r + 8*(L/16), n = L%16
__device__ __forceinline__ v16h frag16(const h16* p0, const h16* p1) {
  v8h a = *(const v8h*)p0, b = *(const v8h*)p1;
  return __builtin_shufflevector(a, b, 0,1,2,3,4,5,6,7,8,9,10,11,12,13,14,15);
}
__device__ __forceinline__ v8f wmma_f16(v16h a, v16h b, v8f c) {
  return __builtin_amdgcn_wmma_f32_16x16x32_f16(false, a, false, b, (short)0, c,
                                                false, false);
}
// ---- fast transcendentals: keep the sequential LSTM cell on TRANS HW ops ----
__device__ __forceinline__ float fexp(float x) {          // e^x
  return __builtin_amdgcn_exp2f(1.4426950408889634f * x); // v_exp_f32
}
__device__ __forceinline__ float fsigm(float x) {         // 1/(1+e^-x)
  return __builtin_amdgcn_rcpf(1.f + __builtin_amdgcn_exp2f(-1.4426950408889634f * x));
}
__device__ __forceinline__ float ftanh(float x) {
#if __has_builtin(__builtin_amdgcn_tanhf)
  return __builtin_amdgcn_tanhf(x);                       // v_tanh_f32 (CDNA5 TRANS)
#else
  float t = __builtin_amdgcn_exp2f(-2.8853900817779268f * fabsf(x)); // e^(-2|x|)
  float r = (1.f - t) * __builtin_amdgcn_rcpf(1.f + t);
  return copysignf(r, x);
#endif
}
__device__ __forceinline__ float hswish(float x) {
  float t = x + 3.f; t = t < 0.f ? 0.f : (t > 6.f ? 6.f : t);
  return x * t * (1.f / 6.f);
}
__device__ __forceinline__ float actf(float x, int ACT) {
  if (ACT == 1) return x > 0.f ? x : 0.f;
  if (ACT == 2) return hswish(x);
  return x;
}

// ---------------- fragment bundle for one 32-K block ------------------------
template<int NT>
struct Frag { v16h a; v16h b[NT]; };

template<int NT>
__device__ __forceinline__ Frag<NT> load_frag(const h16* __restrict__ arow,
                                              const h16* const* wr, int koff) {
  Frag<NT> f;
  f.a = frag16(arow + koff, arow + koff + 16);
#pragma unroll
  for (int nt = 0; nt < NT; nt++)
    f.b[nt] = frag16(wr[nt] + koff, wr[nt] + koff + 16);
  return f;
}
template<int NT>
__device__ __forceinline__ void do_wmma(const Frag<NT>& f, v8f* acc) {
#pragma unroll
  for (int nt = 0; nt < NT; nt++) acc[nt] = wmma_f16(f.a, f.b[nt], acc[nt]);
}

// ---------------- generic WMMA GEMM:  C = act(A @ W^T + bias) ---------------
// A: (M, Kp) f16 row-major.  W: (Np, Kp) f16 row-major.  256 thr = 8 waves,
// block tile 128(M) x 16*NT(N); wave w owns rows [16w,16w+16).
// 2-deep ping/pong pipeline: each fragment set is reloaded into its own
// registers immediately after its WMMAs consume it.
template<int NT, int ACT, bool OUTH>
__global__ __launch_bounds__(256)
void gemm_wmma(const h16* __restrict__ A, const h16* __restrict__ W,
               const float* __restrict__ bias, void* __restrict__ out,
               int M, int Kp, int Np) {
  const int tid = threadIdx.x;
  const int wv = tid >> 5, ln = tid & 31, l = ln & 15, hh = ln >> 4;
  const int m0 = blockIdx.y * 128 + wv * 16;
  const int n0 = blockIdx.x * (16 * NT);
  v8f acc[NT] = {};
  const h16* arow = A + (size_t)(m0 + l) * Kp + 8 * hh;
  const h16* wr[NT];
#pragma unroll
  for (int nt = 0; nt < NT; nt++)
    wr[nt] = W + (size_t)(n0 + nt * 16 + l) * Kp + 8 * hh;

  const int nk = Kp >> 5;                 // #32-wide K blocks (>= 4 for all uses)
  Frag<NT> f0 = load_frag<NT>(arow, wr, 0);
  Frag<NT> f1 = load_frag<NT>(arow, wr, 32);
  int kb = 0;
  for (; kb + 3 < nk; kb += 2) {
    do_wmma<NT>(f0, acc); f0 = load_frag<NT>(arow, wr, (kb + 2) * 32);
    do_wmma<NT>(f1, acc); f1 = load_frag<NT>(arow, wr, (kb + 3) * 32);
  }
  if (kb + 2 < nk) {                      // odd nk: 3 blocks remain
    do_wmma<NT>(f0, acc); f0 = load_frag<NT>(arow, wr, (kb + 2) * 32);
    do_wmma<NT>(f1, acc);
    do_wmma<NT>(f0, acc);
  } else {                                // even nk: 2 blocks remain
    do_wmma<NT>(f0, acc);
    do_wmma<NT>(f1, acc);
  }

#pragma unroll
  for (int nt = 0; nt < NT; nt++) {
    const int n = n0 + nt * 16 + l;
    const float bv = bias ? bias[n] : 0.f;
#pragma unroll
    for (int r = 0; r < 8; r++) {
      const int m = m0 + r + 8 * hh;
      float v = actf(acc[nt][r] + bv, ACT);
      if (OUTH) ((h16*)out)[(size_t)m * Np + n] = (h16)v;
      else      ((float*)out)[(size_t)m * Np + n] = v;
    }
  }
}

// ---------------- grid-wide barrier (generation counter in ws) --------------
__device__ __forceinline__ void grid_sync(unsigned* sync, unsigned nwg) {
  __syncthreads();
  if (threadIdx.x == 0) {
    __threadfence();
    unsigned* cnt = sync;
    unsigned* gen = sync + 1;
    unsigned g = __hip_atomic_load(gen, __ATOMIC_RELAXED, __HIP_MEMORY_SCOPE_AGENT);
    unsigned a = __hip_atomic_fetch_add(cnt, 1u, __ATOMIC_ACQ_REL, __HIP_MEMORY_SCOPE_AGENT);
    if (a == nwg - 1u) {
      __hip_atomic_store(cnt, 0u, __ATOMIC_RELAXED, __HIP_MEMORY_SCOPE_AGENT);
      __hip_atomic_fetch_add(gen, 1u, __ATOMIC_RELEASE, __HIP_MEMORY_SCOPE_AGENT);
    } else {
      while (__hip_atomic_load(gen, __ATOMIC_ACQUIRE, __HIP_MEMORY_SCOPE_AGENT) == g)
        __builtin_amdgcn_s_sleep(1);
    }
    __threadfence();
  }
  __syncthreads();
}

// gates = [X0 | X1] @ Wcat^T for this lane's 4 gate-aligned 16x16 tiles.
// B-fragments come from LDS (weights resident for the whole recurrence):
// lds row (layer*4+s)*16 + l holds global weight row s*544 + 16*wg + l.
__device__ __forceinline__ Frag<4> load_gate_frag(const h16* __restrict__ a0,
                                                  const h16* __restrict__ a1,
                                                  const h16* lb, int kb) {
  const h16* ar = (kb < 17) ? (a0 + kb * 32) : (a1 + (kb - 17) * 32);
  Frag<4> f;
  f.a = frag16(ar, ar + 16);
#pragma unroll
  for (int s = 0; s < 4; ++s) {
    const h16* p = lb + s * (16 * LDS_ROW) + kb * 32;
    f.b[s] = frag16(p, p + 16);                      // ds_read_b128 x2
  }
  return f;
}
__device__ __forceinline__ void gate_gemm(const h16* __restrict__ X0,
                                          const h16* __restrict__ X1,
                                          const h16* lb,
                                          int mrow, int hh, v8f acc[4]) {
  const h16* a0 = X0 + (size_t)mrow * CORE_P + 8 * hh;
  const h16* a1 = X1 + (size_t)mrow * CORE_P + 8 * hh;
  Frag<4> f0 = load_gate_frag(a0, a1, lb, 0);
  Frag<4> f1 = load_gate_frag(a0, a1, lb, 1);
  int kb = 0;
  for (; kb + 3 < 34; kb += 2) {          // 34 blocks total (even)
    do_wmma<4>(f0, acc); f0 = load_gate_frag(a0, a1, lb, kb + 2);
    do_wmma<4>(f1, acc); f1 = load_gate_frag(a0, a1, lb, kb + 3);
  }
  do_wmma<4>(f0, acc);                    // kb == 32
  do_wmma<4>(f1, acc);                    // kb == 33
}

// ---------------- persistent LSTM kernel (34 WGs, 512 steps) ----------------
// h1u: unmasked h1 (written phase1, read phase2)
// h1m: masked h1   (written phase2, read next phase1)
// h2m0/h2m1: masked h2 ping-pong by step parity
// c1/c2 live entirely in registers. 2 grid barriers per step.
// Dynamic LDS: 128 rows x LDS_ROW h16 = 274KB (both layers' weight slices).
__global__ __launch_bounds__(256)
void lstm_kernel(const h16* __restrict__ CI,
                 const h16* __restrict__ Wc0, const h16* __restrict__ Wc1,
                 const float* __restrict__ bg0, const float* __restrict__ bg1,
                 const unsigned char* __restrict__ done,
                 h16* __restrict__ h1u, h16* __restrict__ h1m,
                 h16* __restrict__ h2m0, h16* __restrict__ h2m1,
                 h16* __restrict__ H2, unsigned* __restrict__ sync) {
  extern __shared__ h16 smem[];           // [128][LDS_ROW]
  const int tid = threadIdx.x, wg = blockIdx.x;
  const int wv = tid >> 5, ln = tid & 31, l = ln & 15, hh = ln >> 4;
  const int kc = wg * 16 + l;        // gate column 0..543 owned by this lane
  const int mrow = wv * 16 + l;      // A-fragment row

  // ---- one-time fill: this WG's 64 weight rows per layer into LDS ----
  // 128 rows x 136 v8h chunks = 17408 chunks, 68 per thread.
  for (int cid = tid; cid < 128 * 136; cid += 256) {
    const int row = cid / 136, c8 = cid % 136;
    const int layer = row >> 6, within = row & 63;
    const int s = within >> 4, r = within & 15;
    const h16* src = (layer ? Wc1 : Wc0) +
                     (size_t)(s * CORE_P + wg * 16 + r) * KCAT + c8 * 8;
    *(v8h*)(smem + (size_t)row * LDS_ROW + c8 * 8) = *(const v8h*)src;
  }
  __syncthreads();
  const h16* lb0 = smem + (size_t)l * LDS_ROW + 8 * hh;           // layer-0 base
  const h16* lb1 = smem + (size_t)(64 + l) * LDS_ROW + 8 * hh;    // layer-1 base

  float bi0[4], bi1[4];
#pragma unroll
  for (int s = 0; s < 4; ++s) { bi0[s] = bg0[s * CORE_P + kc]; bi1[s] = bg1[s * CORE_P + kc]; }
  float c1[8] = {}, c2[8] = {}, h1v[8];

  for (int t = 0; t < T_STEPS; ++t) {
    const h16* h2r = (t & 1) ? h2m1 : h2m0;
    h16* h2w = (t & 1) ? h2m0 : h2m1;
    // ---- layer 1: gates = [CI_t | h1(masked)] @ Wc0^T ----
    {
      v8f acc[4] = {};
      gate_gemm(CI + (size_t)t * BATCH * CORE_P, h1m, lb0, mrow, hh, acc);
#pragma unroll
      for (int r = 0; r < 8; ++r) {
        float gi = fsigm(acc[0][r] + bi0[0]);
        float gf = fsigm(acc[1][r] + bi0[1]);
        float gg = ftanh(acc[2][r] + bi0[2]);
        float go = fsigm(acc[3][r] + bi0[3]);
        float cn = gf * c1[r] + gi * gg;
        float hn = go * ftanh(cn);
        c1[r] = cn; h1v[r] = hn;
        const int row = wv * 16 + r + 8 * hh;
        h1u[(size_t)row * CORE_P + kc] = (h16)hn;
      }
    }
    grid_sync(sync, NWG_LSTM);
    // ---- layer 2: gates = [h1(new) | h2(masked,prev)] @ Wc1^T ----
    {
      // prefetch next step's CI rows into cache (global_prefetch_b8)
      if (t + 1 < T_STEPS && hh == 0) {
        const h16* nrow = CI + (size_t)(t + 1) * BATCH * CORE_P + (size_t)mrow * CORE_P;
#pragma unroll
        for (int p = 0; p < 5; ++p) __builtin_prefetch(nrow + p * 128, 0, 3);
      }
      v8f acc[4] = {};
      gate_gemm(h1u, h2r, lb1, mrow, hh, acc);
#pragma unroll
      for (int r = 0; r < 8; ++r) {
        float gi = fsigm(acc[0][r] + bi1[0]);
        float gf = fsigm(acc[1][r] + bi1[1]);
        float gg = ftanh(acc[2][r] + bi1[2]);
        float go = fsigm(acc[3][r] + bi1[3]);
        float cn = gf * c2[r] + gi * gg;
        float hn = go * ftanh(cn);
        const int row = wv * 16 + r + 8 * hh;
        H2[(size_t)t * BATCH * CORE_P + (size_t)row * CORE_P + kc] = (h16)hn; // out
        const float nd = done[t * BATCH + row] ? 0.f : 1.f;                  // mask carry
        c2[r] = cn * nd;
        c1[r] *= nd;
        h2w[(size_t)row * CORE_P + kc] = (h16)(hn * nd);
        h1m[(size_t)row * CORE_P + kc] = (h16)(h1v[r] * nd);
      }
    }
    grid_sync(sync, NWG_LSTM);
  }
}

// ---------------- elementwise / packing kernels -----------------------------
__global__ void k_cvt_f16(h16* dst, const float* src, int n) {
  int i = blockIdx.x * 256 + threadIdx.x;
  if (i < n) dst[i] = (h16)src[i];
}
// zero-padded (srcN,srcK) f32 -> (Np,Kp) f16
__global__ void k_pack(h16* dst, const float* src, int srcN, int srcK, int Np, int Kp) {
  int i = blockIdx.x * 256 + threadIdx.x;
  if (i >= Np * Kp) return;
  int r = i / Kp, c = i % Kp;
  dst[i] = (h16)((r < srcN && c < srcK) ? src[(size_t)r * srcK + c] : 0.f);
}
// Wcat (G4P, KCAT): rows = 4 sections of CORE_P (per-gate padding),
// cols = [Wih K-padded | Whh K-padded]
__global__ void k_pack_gate(h16* dst, const float* Wih, const float* Whh) {
  int i = blockIdx.x * 256 + threadIdx.x;
  if (i >= G4P * KCAT) return;
  int r = i / KCAT, c = i % KCAT;
  int s = r / CORE_P, rr = r % CORE_P;
  float v = 0.f;
  if (rr < CORE_R) {
    size_t sr = (size_t)(s * CORE_R + rr) * CORE_R;
    if (c < CORE_R) v = Wih[sr + c];
    else if (c >= CORE_P && (c - CORE_P) < CORE_R) v = Whh[sr + (c - CORE_P)];
  }
  dst[i] = (h16)v;
}
__global__ void k_pack_gbias(float* dst, const float* bih, const float* bhh) {
  int i = blockIdx.x * 256 + threadIdx.x;
  if (i >= G4P) return;
  int s = i / CORE_P, rr = i % CORE_P;
  dst[i] = (rr < CORE_R) ? (bih[s * CORE_R + rr] + bhh[s * CORE_R + rr]) : 0.f;
}
__global__ void k_pack_b32(float* dst, const float* src, int n) {
  int i = blockIdx.x * 256 + threadIdx.x;
  if (i < 32) dst[i] = (i < n) ? src[i] : 0.f;
}
// core_input = [E1(512) | clip(reward) | last_action(30) | 0-pad]
__global__ void k_build_ci(h16* CI, const h16* E1, const float* reward,
                           const float* la) {
  int i = blockIdx.x * 256 + threadIdx.x;
  if (i >= ROWS * CORE_P) return;
  int row = i / CORE_P, col = i % CORE_P;
  float v;
  if (col < NDIM)        v = (float)E1[(size_t)row * NDIM + col];
  else if (col == NDIM)  { v = reward[row]; v = fminf(fmaxf(v, -1.f), 1.f); }
  else if (col < CORE_R) v = la[(size_t)row * ADIM + (col - NDIM - 1)];
  else                   v = 0.f;
  CI[i] = (h16)v;
}
// Yr = relu(H2); H2 <- hardswish(H2) in place
__global__ void k_prep_heads(h16* Yr, h16* H2) {
  int i = blockIdx.x * 256 + threadIdx.x;
  if (i >= ROWS * CORE_P) return;
  float v = (float)H2[i];
  Yr[i] = (h16)(v > 0.f ? v : 0.f);
  H2[i] = (h16)hswish(v);
}
// d_out = [policy_logits (ROWS,60) | baseline (ROWS) | action (ROWS,30)]
__global__ void k_final(const float* AM, const float* LS, const h16* B0,
                        const float* Wb1, const float* bb1, float* out) {
  int row = blockIdx.x * 256 + threadIdx.x;
  if (row >= ROWS) return;
  float bl = bb1[0];
#pragma unroll
  for (int j = 0; j < 32; ++j) bl += (float)B0[(size_t)row * 32 + j] * Wb1[j];
  float* PL = out;
  float* BL = out + (size_t)ROWS * 60;
  float* AC = BL + ROWS;
  BL[row] = bl;
#pragma unroll
  for (int j = 0; j < ADIM; ++j) {
    float am = AM[(size_t)row * 32 + j];
    float ls = LS[(size_t)row * 32 + j];
    ls = fminf(fmaxf(ls, -20.f), 2.f);
    PL[(size_t)row * 60 + j] = am;
    PL[(size_t)row * 60 + 30 + j] = fexp(ls);
    AC[(size_t)row * 30 + j] = ftanh(am);
  }
}

// ---------------------------------------------------------------------------
extern "C" void kernel_launch(void* const* d_in, const int* in_sizes, int n_in,
                              void* d_out, int out_size, void* d_ws, size_t ws_size,
                              hipStream_t stream) {
  (void)in_sizes; (void)n_in; (void)out_size; (void)ws_size;
  const float* frame  = (const float*)d_in[0];
  const float* reward = (const float*)d_in[1];
  const float* la     = (const float*)d_in[2];
  const unsigned char* done = (const unsigned char*)d_in[3]; // numpy bool = 1 byte
  const float* Ws0 = (const float*)d_in[4];
  const float* bs0 = (const float*)d_in[5];
  const float* Ws1 = (const float*)d_in[6];
  const float* bs1 = (const float*)d_in[7];
  const float* Wih0 = (const float*)d_in[8];
  const float* Whh0 = (const float*)d_in[9];
  const float* bih0 = (const float*)d_in[10];
  const float* bhh0 = (const float*)d_in[11];
  const float* Wih1 = (const float*)d_in[12];
  const float* Whh1 = (const float*)d_in[13];
  const float* bih1 = (const float*)d_in[14];
  const float* bhh1 = (const float*)d_in[15];
  const float* Wp0 = (const float*)d_in[16];
  const float* bp0 = (const float*)d_in[17];
  const float* Wp1 = (const float*)d_in[18];
  const float* bp1 = (const float*)d_in[19];
  const float* Wls = (const float*)d_in[20];
  const float* bls = (const float*)d_in[21];
  const float* Wb0 = (const float*)d_in[22];
  const float* bb0 = (const float*)d_in[23];
  const float* Wb1 = (const float*)d_in[24];
  const float* bb1 = (const float*)d_in[25];

  // ---- workspace layout (assumes ws_size >= ~310 MB) ----
  char* ws = (char*)d_ws;
  size_t off = 0;
  auto alloc = [&](size_t b) { size_t r = off; off += (b + 255) & ~(size_t)255; return r; };
  const size_t HB = (size_t)BATCH * CORE_P * 2;                 // 139264 B
  size_t oH1U = alloc(HB), oH1M = alloc(HB);
  size_t oH2A = alloc(HB), oH2B = alloc(HB);
  size_t oSync = alloc(256);
  size_t oF16 = alloc((size_t)ROWS * SDIM * 2);
  size_t oE0  = alloc((size_t)ROWS * NDIM * 2);
  size_t oE1  = alloc((size_t)ROWS * NDIM * 2);
  size_t oCI  = alloc((size_t)ROWS * CORE_P * 2);
  size_t oWs0 = alloc((size_t)NDIM * SDIM * 2);
  size_t oWs1 = alloc((size_t)NDIM * NDIM * 2);
  size_t oWc0 = alloc((size_t)G4P * KCAT * 2);
  size_t oWc1 = alloc((size_t)G4P * KCAT * 2);
  size_t oWp0 = alloc((size_t)NDIM * CORE_P * 2);
  size_t oWp1 = alloc((size_t)32 * NDIM * 2);
  size_t oWls = alloc((size_t)32 * CORE_P * 2);
  size_t oWb0 = alloc((size_t)32 * CORE_P * 2);
  size_t obg0 = alloc(G4P * 4), obg1 = alloc(G4P * 4);
  size_t obp1 = alloc(128), obls = alloc(128);
  size_t oH2  = alloc((size_t)ROWS * CORE_P * 2);
  // aliases (lifetimes disjoint): Yr reuses CI; Z0 reuses E0; AM/LS/B0 reuse E1
  size_t oYr = oCI, oZ0 = oE0;
  size_t oAM = oE1, oLS = oE1 + (size_t)ROWS * 32 * 4, oB0 = oLS + (size_t)ROWS * 32 * 4;

  h16* F16 = (h16*)(ws + oF16); h16* E0 = (h16*)(ws + oE0); h16* E1 = (h16*)(ws + oE1);
  h16* CI = (h16*)(ws + oCI);  h16* H2 = (h16*)(ws + oH2);
  unsigned* sync = (unsigned*)(ws + oSync);

  // zero h-state + sync counters (contiguous block)
  hipMemsetAsync(ws + oH1U, 0, 4 * HB + 256, stream);

  // ---- pack weights to padded f16 ----
  auto nb = [](size_t n) { return (unsigned)((n + 255) / 256); };
  k_cvt_f16<<<nb((size_t)ROWS * SDIM), 256, 0, stream>>>(F16, frame, ROWS * SDIM);
  k_pack<<<nb((size_t)NDIM * SDIM), 256, 0, stream>>>((h16*)(ws + oWs0), Ws0, NDIM, SDIM, NDIM, SDIM);
  k_pack<<<nb((size_t)NDIM * NDIM), 256, 0, stream>>>((h16*)(ws + oWs1), Ws1, NDIM, NDIM, NDIM, NDIM);
  k_pack_gate<<<nb((size_t)G4P * KCAT), 256, 0, stream>>>((h16*)(ws + oWc0), Wih0, Whh0);
  k_pack_gate<<<nb((size_t)G4P * KCAT), 256, 0, stream>>>((h16*)(ws + oWc1), Wih1, Whh1);
  k_pack_gbias<<<nb(G4P), 256, 0, stream>>>((float*)(ws + obg0), bih0, bhh0);
  k_pack_gbias<<<nb(G4P), 256, 0, stream>>>((float*)(ws + obg1), bih1, bhh1);
  k_pack<<<nb((size_t)NDIM * CORE_P), 256, 0, stream>>>((h16*)(ws + oWp0), Wp0, NDIM, CORE_R, NDIM, CORE_P);
  k_pack<<<nb((size_t)32 * NDIM), 256, 0, stream>>>((h16*)(ws + oWp1), Wp1, ADIM, NDIM, 32, NDIM);
  k_pack<<<nb((size_t)32 * CORE_P), 256, 0, stream>>>((h16*)(ws + oWls), Wls, ADIM, CORE_R, 32, CORE_P);
  k_pack<<<nb((size_t)32 * CORE_P), 256, 0, stream>>>((h16*)(ws + oWb0), Wb0, 32, CORE_R, 32, CORE_P);
  k_pack_b32<<<1, 256, 0, stream>>>((float*)(ws + obp1), bp1, ADIM);
  k_pack_b32<<<1, 256, 0, stream>>>((float*)(ws + obls), bls, ADIM);

  // ---- encoder: E0 = relu(F@Ws0^T+bs0), E1 = relu(E0@Ws1^T+bs1) ----
  gemm_wmma<4, 1, true><<<dim3(NDIM / 64, ROWS / 128), 256, 0, stream>>>(
      F16, (h16*)(ws + oWs0), bs0, E0, ROWS, SDIM, NDIM);
  gemm_wmma<4, 1, true><<<dim3(NDIM / 64, ROWS / 128), 256, 0, stream>>>(
      E0, (h16*)(ws + oWs1), bs1, E1, ROWS, NDIM, NDIM);
  k_build_ci<<<nb((size_t)ROWS * CORE_P), 256, 0, stream>>>(CI, E1, reward, la);

  // ---- sequential LSTM (persistent, grid-synced, weights in LDS) ----
  lstm_kernel<<<NWG_LSTM, 256, (size_t)128 * LDS_ROW * 2, stream>>>(
      CI, (h16*)(ws + oWc0), (h16*)(ws + oWc1),
      (float*)(ws + obg0), (float*)(ws + obg1), done,
      (h16*)(ws + oH1U), (h16*)(ws + oH1M),
      (h16*)(ws + oH2A), (h16*)(ws + oH2B), H2, sync);

  // ---- heads ----
  k_prep_heads<<<nb((size_t)ROWS * CORE_P), 256, 0, stream>>>((h16*)(ws + oYr), H2);
  // Z0 = hardswish(relu(y)@Wp0^T + bp0)
  gemm_wmma<4, 2, true><<<dim3(NDIM / 64, ROWS / 128), 256, 0, stream>>>(
      (h16*)(ws + oYr), (h16*)(ws + oWp0), bp0, (h16*)(ws + oZ0), ROWS, CORE_P, NDIM);
  // AM = Z0 @ Wp1^T + bp1
  gemm_wmma<2, 0, false><<<dim3(1, ROWS / 128), 256, 0, stream>>>(
      (h16*)(ws + oZ0), (h16*)(ws + oWp1), (float*)(ws + obp1), (float*)(ws + oAM), ROWS, NDIM, 32);
  // LS = hardswish(y) @ Wls^T + bls   (H2 now holds hardswish(y))
  gemm_wmma<2, 0, false><<<dim3(1, ROWS / 128), 256, 0, stream>>>(
      H2, (h16*)(ws + oWls), (float*)(ws + obls), (float*)(ws + oLS), ROWS, CORE_P, 32);
  // B0 = relu(relu(y) @ Wb0^T + bb0)
  gemm_wmma<2, 1, true><<<dim3(1, ROWS / 128), 256, 0, stream>>>(
      (h16*)(ws + oYr), (h16*)(ws + oWb0), bb0, (h16*)(ws + oB0), ROWS, CORE_P, 32);
  // outputs
  k_final<<<nb(ROWS), 256, 0, stream>>>((float*)(ws + oAM), (float*)(ws + oLS),
                                        (h16*)(ws + oB0), Wb1, bb1, (float*)d_out);
}
